// CondNet_8735963480705
// MI455X (gfx1250) — matrix operand order repaired and checked by reading, not compiled, not run on hardware
//
#include <hip/hip_runtime.h>
#include <hip/hip_bf16.h>
#include <math.h>

// ---------------------------------------------------------------------------
// CondNet fused kernel for MI455X (gfx1250, wave32, WMMA 16x16x32 f16->f32)
// One wave handles 16 batch rows end-to-end. All D=64 contractions are WMMA.
// x tile staged into LDS via the Tensor Data Mover (TENSORcnt) when available.
// ---------------------------------------------------------------------------

typedef __attribute__((ext_vector_type(16))) _Float16 v16h;
typedef __attribute__((ext_vector_type(8)))  float    v8f;
typedef unsigned int u32x4 __attribute__((ext_vector_type(4)));
typedef int          i32x4 __attribute__((ext_vector_type(4)));
typedef int          i32x8 __attribute__((ext_vector_type(8)));

#if __has_builtin(__builtin_amdgcn_tensor_load_to_lds) && __has_builtin(__builtin_amdgcn_s_wait_tensorcnt)
#define HAVE_TDM 1
#else
#define HAVE_TDM 0
#endif

struct CondParams {
  const float* x;
  const float *inn_w, *inn_b, *jet_w, *jet_b, *dil_w, *dil_b, *ang_w, *ang_b;
  const float *s_ln_g, *s_ln_b, *s_in_w, *s_in_b, *s_out_w, *s_out_b;
  const float *s_fln_g, *s_fln_b, *s_f_w, *s_f_b;
  const float *c_lnq_g, *c_lnq_b, *c_lnk_g, *c_lnk_b, *c_in_w, *c_in_b;
  const float *c_out_w, *c_out_b, *c_fln_g, *c_fln_b, *c_f_w, *c_f_b;
  const float *h_ln_g, *h_ln_b, *h_w, *h_b;
  float* out;
  long long B;
};

// ----- LDS budget ----------------------------------------------------------
// weights (f16): qkv_s 12288 | out_s 4096 | ffn_s 4096 | qkv_c 12288
//                out_c 4096  | ffn_c 4096 | head 2048          = 43008 h = 86016 B
#define W_HALVES_QKV_S 0
#define W_HALVES_O_S   12288
#define W_HALVES_F_S   16384
#define W_HALVES_QKV_C 20480
#define W_HALVES_O_C   32768
#define W_HALVES_F_C   36864
#define W_HALVES_HEAD  40960
#define W_BYTES        86016
// per-wave: TOK f32[6][16][64]=24576B | TMP f16[5][16][64]=10240B
//           QKV f16[5][16][192]=30720B | XR f32[16][37]=2368B(+pad) | MSK f32[16][8]=512B
#define XR_OFF         65536
#define MSK_OFF        68096
#define WAVE_BYTES     68608
#define N_WAVES        3
#define SMEM_TOTAL     (W_BYTES + N_WAVES * WAVE_BYTES)   // 291840 B <= 320KB

// ----- WMMA fragment loaders (ISA 7.12.2 layouts, wave32) ------------------
// A: 16x32 f16, lane = h*16+m holds row m, K in {8h..8h+7} U {16+8h..16+8h+7}
__device__ __forceinline__ v16h ldA(const _Float16* A /*[16][64]*/, int k0, int lane) {
  const int m = lane & 15, h = lane >> 4;
  v16h a;
#pragma unroll
  for (int v = 0; v < 8; ++v) {
    const int kb = ((v & 4) ? 16 : 0) + 8 * h + 2 * (v & 3);
    const _Float16* p = A + m * 64 + k0 + kb;
    a[2 * v]     = p[0];
    a[2 * v + 1] = p[1];
  }
  return a;
}

// B: 32x16 f16 from W[N][64] (row-major over K). lane = h*16+n holds
// K in {8h..8h+7} U {16+8h..16+8h+7}; column n = n0 + lane%16.
__device__ __forceinline__ v16h ldB(const _Float16* W /*[N][64]*/, int n0, int k0, int lane) {
  const int n = n0 + (lane & 15), h = lane >> 4;
  const _Float16* p = W + n * 64 + k0 + 8 * h;
  v16h b;
#pragma unroll
  for (int v = 0; v < 8; ++v) {
    const int kb = ((v & 4) ? 8 : 0) + 2 * v;
    b[2 * v]     = p[kb];
    b[2 * v + 1] = p[kb + 1];
  }
  return b;
}

// C[16xN] = A[16x64] @ W[N][64]^T + bias ; either accumulate into f32 LDS
// (residual) or write f16 LDS (activation staging).
__device__ __forceinline__ void gemm16x64(const _Float16* A, const _Float16* W,
                                          const float* bias, int Nout, int lane,
                                          float* oF, int ldF,
                                          _Float16* oH, int ldH) {
  for (int n0 = 0; n0 < Nout; n0 += 16) {
    v8f acc = {};
    {
      v16h a = ldA(A, 0, lane);
      v16h b = ldB(W, n0, 0, lane);
      acc = __builtin_amdgcn_wmma_f32_16x16x32_f16(false, a, false, b, (short)0, acc, false, false);
    }
    {
      v16h a = ldA(A, 32, lane);
      v16h b = ldB(W, n0, 32, lane);
      acc = __builtin_amdgcn_wmma_f32_16x16x32_f16(false, a, false, b, (short)0, acc, false, false);
    }
    const int   n  = n0 + (lane & 15);
    const float bv = bias[n];
    const int   mh = (lane >> 4) * 8;
#pragma unroll
    for (int r = 0; r < 8; ++r) {
      const float val = acc[r] + bv;
      if (oF) oF[(mh + r) * ldF + n] += val;
      if (oH) oH[(mh + r) * ldH + n] = (_Float16)val;
    }
  }
}

// layernorm (+ optional exact gelu) over 64 channels, f32 -> f16
__device__ __forceinline__ void ln_to_f16(const float* x64, _Float16* y64,
                                          const float* g, const float* b, bool gelu) {
  float m = 0.f;
#pragma unroll 8
  for (int i = 0; i < 64; ++i) m += x64[i];
  m *= (1.f / 64.f);
  float v = 0.f;
#pragma unroll 8
  for (int i = 0; i < 64; ++i) { const float d = x64[i] - m; v += d * d; }
  const float rs = rsqrtf(v * (1.f / 64.f) + 1e-5f);
#pragma unroll 8
  for (int i = 0; i < 64; ++i) {
    float t = (x64[i] - m) * rs * g[i] + b[i];
    if (gelu) t = 0.5f * t * (1.f + erff(t * 0.70710678118f));
    y64[i] = (_Float16)t;
  }
}

__device__ __forceinline__ void stage_w(const float* src, _Float16* dst, int n,
                                        int tid, int nth) {
  for (int i = tid; i < n; i += nth) dst[i] = (_Float16)src[i];
}

__global__ __launch_bounds__(96)
void condnet_fused(CondParams P) {
  extern __shared__ char smem[];
  const int tid  = threadIdx.x;
  const int lane = tid & 31;
  const int wave = tid >> 5;

  _Float16* WH = (_Float16*)smem;
  _Float16* Wqkv_s = WH + W_HALVES_QKV_S;
  _Float16* Wo_s   = WH + W_HALVES_O_S;
  _Float16* Wf_s   = WH + W_HALVES_F_S;
  _Float16* Wqkv_c = WH + W_HALVES_QKV_C;
  _Float16* Wo_c   = WH + W_HALVES_O_C;
  _Float16* Wf_c   = WH + W_HALVES_F_C;
  _Float16* Whd    = WH + W_HALVES_HEAD;

  char* wb = smem + W_BYTES + wave * WAVE_BYTES;
  float*    TOK = (float*)wb;                     // [6][16][64]
  _Float16* TMP = (_Float16*)(wb + 24576);        // [5][16][64]
  _Float16* QKV = (_Float16*)(wb + 34816);        // [5][16][192] (q|k|v)
  float*    XR  = (float*)(wb + XR_OFF);          // [16][37] dense
  float*    MSK = (float*)(wb + MSK_OFF);         // [16][8]

  const long long rowBase = (long long)blockIdx.x * (16 * N_WAVES) + (long long)wave * 16;

  // ---- kick off x-tile DMA into LDS via the Tensor Data Mover ----
  // The 16x37-f32 tile is contiguous in HBM (x is [B][37] dense): one 1-D
  // tensor_load_to_lds per wave, zero-fill past the end of x (tail blocks).
#if HAVE_TDM
  {
    const unsigned long long ga =
        (unsigned long long)(const void*)P.x + (unsigned long long)rowBase * 148ull;
    long long rem = P.B - rowBase;
    if (rem < 0) rem = 0;
    const unsigned int td0 = (unsigned int)(rem * 37ll);       // elements until end of x
    const unsigned int ldsOff = (unsigned int)(W_BYTES + wave * WAVE_BYTES + XR_OFF);
    u32x4 g0 = {0, 0, 0, 0};
    i32x8 g1 = {0, 0, 0, 0, 0, 0, 0, 0};
    i32x4 g2 = {0, 0, 0, 0};
    i32x4 g3 = {0, 0, 0, 0};
    g0[0] = 1u;                                                // count=1, user mode
    g0[1] = ldsOff;                                            // lds_addr (bytes)
    g0[2] = (unsigned int)(ga & 0xffffffffull);                // global_addr[31:0]
    g0[3] = (unsigned int)((ga >> 32) & 0x01ffffffull) | 0x80000000u; // [56:32] | type=2
    g1[0] = (int)0x00020000u;                                  // data_size = 4B
    g1[1] = (int)((td0 & 0xffffu) << 16);                      // tensor_dim0[15:0]
    g1[2] = (int)((td0 >> 16) | (1u << 16));                   // tensor_dim0[31:16] | tensor_dim1=1
    g1[3] = (int)(592u << 16);                                 // tile_dim0 = 16*37 elements
    g1[5] = 592;                                               // tensor_dim0_stride (unused, 1-D)
#if __clang_major__ >= 23
    i32x8 g4 = {0, 0, 0, 0, 0, 0, 0, 0};
    __builtin_amdgcn_tensor_load_to_lds(g0, g1, g2, g3, g4, 0);
#else
    __builtin_amdgcn_tensor_load_to_lds(g0, g1, g2, g3, 0);
#endif
  }
#endif

  // ---- stage weights f32 -> f16 into LDS (whole block) ----
  stage_w(P.s_in_w,  Wqkv_s, 192 * 64, tid, 96);
  stage_w(P.s_out_w, Wo_s,    64 * 64, tid, 96);
  stage_w(P.s_f_w,   Wf_s,    64 * 64, tid, 96);
  stage_w(P.c_in_w,  Wqkv_c, 192 * 64, tid, 96);
  stage_w(P.c_out_w, Wo_c,    64 * 64, tid, 96);
  stage_w(P.c_f_w,   Wf_c,    64 * 64, tid, 96);
  stage_w(P.h_w,     Whd,     32 * 64, tid, 96);

#if HAVE_TDM
  __builtin_amdgcn_s_wait_tensorcnt(0);   // x tile resident in LDS
#else
  for (int i = lane; i < 16 * 37; i += 32) {
    const int m = i / 37, c = i % 37;
    long long r = rowBase + m;
    if (r >= P.B) r = P.B - 1;
    XR[m * 37 + c] = P.x[r * 37 + c];
  }
#endif
  __syncthreads();   // weights staged (and per-wave x tile visible)

  // ---- key mask per row: |cond[4t:4t+4]| sum == 0 for t<3 ----
  if (lane < 16) {
    const float* xr = XR + lane * 37;
#pragma unroll
    for (int t = 0; t < 5; ++t) {
      float s = 0.f;
      if (t < 3)
        for (int k = 0; k < 4; ++k) s += fabsf(xr[16 + 4 * t + k]);
      MSK[lane * 8 + t] = (t < 3 && s == 0.f) ? 1.f : 0.f;
    }
  }

  // ---- input projections -> TOK (VALU; K<=16) ----
  for (int it = lane; it < 16 * 6 * 64; it += 32) {
    const int m = it & 15;
    const int r = it >> 4;
    const int t = r % 6;
    const int o = r / 6;
    const float* w; const float* bb; int xo, dn;
    switch (t) {
      case 0: w = P.jet_w; bb = P.jet_b; xo = 16; dn = 4;  break;
      case 1: w = P.jet_w; bb = P.jet_b; xo = 20; dn = 4;  break;
      case 2: w = P.jet_w; bb = P.jet_b; xo = 24; dn = 4;  break;
      case 3: w = P.dil_w; bb = P.dil_b; xo = 28; dn = 4;  break;
      case 4: w = P.ang_w; bb = P.ang_b; xo = 32; dn = 5;  break;
      default: w = P.inn_w; bb = P.inn_b; xo = 0; dn = 16; break;
    }
    float acc = bb[o];
    const float* xr = XR + m * 37 + xo;
    for (int k = 0; k < dn; ++k) acc += xr[k] * w[o * dn + k];
    TOK[(t * 16 + m) * 64 + o] = acc;
  }

  // ================= SELF BLOCK (5 context tokens) =================
  // ln(context) -> TMP
  for (int it = lane; it < 80; it += 32) {
    const int t = it / 16, m = it % 16;
    ln_to_f16(TOK + (t * 16 + m) * 64, TMP + (t * 16 + m) * 64, P.s_ln_g, P.s_ln_b, false);
  }
  // qkv = ln(c) @ Wqkv^T + b  -> QKV (f16)
  for (int t = 0; t < 5; ++t)
    gemm16x64(TMP + t * 1024, Wqkv_s, P.s_in_b, 192, lane,
              nullptr, 0, QKV + t * 3072, 192);
  // attention (HD=8, 5 keys) -> TMP
  for (int it = lane; it < 128; it += 32) {
    const int m = it & 15, h = it >> 4;
    for (int qt = 0; qt < 5; ++qt) {
      float sc[5]; float mx = -1e30f;
      for (int kt = 0; kt < 5; ++kt) {
        const _Float16* qv = QKV + qt * 3072 + m * 192 + h * 8;
        const _Float16* kv = QKV + kt * 3072 + m * 192 + 64 + h * 8;
        float s = 0.f;
#pragma unroll
        for (int d = 0; d < 8; ++d) s += (float)qv[d] * (float)kv[d];
        s *= 0.35355339059f;
        if (MSK[m * 8 + kt] != 0.f) s = -1e9f;
        sc[kt] = s; mx = fmaxf(mx, s);
      }
      float den = 0.f;
      for (int kt = 0; kt < 5; ++kt) { sc[kt] = expf(sc[kt] - mx); den += sc[kt]; }
      const float inv = 1.f / den;
#pragma unroll
      for (int d = 0; d < 8; ++d) {
        float o = 0.f;
        for (int kt = 0; kt < 5; ++kt)
          o += sc[kt] * (float)QKV[kt * 3072 + m * 192 + 128 + h * 8 + d];
        TMP[qt * 1024 + m * 64 + h * 8 + d] = (_Float16)(o * inv);
      }
    }
  }
  // out-proj + residual into TOK
  for (int t = 0; t < 5; ++t)
    gemm16x64(TMP + t * 1024, Wo_s, P.s_out_b, 64, lane,
              TOK + t * 1024, 64, nullptr, 0);
  // ffn: gelu(ln(c)) -> TMP; TOK += TMP @ Wf^T + b
  for (int it = lane; it < 80; it += 32) {
    const int t = it / 16, m = it % 16;
    ln_to_f16(TOK + (t * 16 + m) * 64, TMP + (t * 16 + m) * 64, P.s_fln_g, P.s_fln_b, true);
  }
  for (int t = 0; t < 5; ++t)
    gemm16x64(TMP + t * 1024, Wf_s, P.s_f_b, 64, lane,
              TOK + t * 1024, 64, nullptr, 0);
  // c = where(mask, 0, c)
  for (int it = lane; it < 80; it += 32) {
    const int t = it / 16, m = it % 16;
    if (MSK[m * 8 + t] != 0.f)
      for (int i = 0; i < 64; ++i) TOK[(t * 16 + m) * 64 + i] = 0.f;
  }

  // ================= CROSS BLOCK (query token = TOK[5]) =================
  // q = ln_q(query) -> TMP[0]; project to QKV[0][:,0:64]
  for (int it = lane; it < 16; it += 32)
    ln_to_f16(TOK + (5 * 16 + it) * 64, TMP + it * 64, P.c_lnq_g, P.c_lnq_b, false);
  gemm16x64(TMP, Wqkv_c, P.c_in_b, 64, lane, nullptr, 0, QKV, 192);
  // kv = ln_kv(c) -> TMP[t]; project to QKV[t][:,64:192]
  for (int it = lane; it < 80; it += 32) {
    const int t = it / 16, m = it % 16;
    ln_to_f16(TOK + (t * 16 + m) * 64, TMP + (t * 16 + m) * 64, P.c_lnk_g, P.c_lnk_b, false);
  }
  for (int t = 0; t < 5; ++t)
    gemm16x64(TMP + t * 1024, Wqkv_c + 64 * 64, P.c_in_b + 64, 128, lane,
              nullptr, 0, QKV + t * 3072 + 64, 192);
  // cross attention (single query token) -> TMP[0]
  for (int it = lane; it < 128; it += 32) {
    const int m = it & 15, h = it >> 4;
    const _Float16* qv = QKV + m * 192 + h * 8;
    float sc[5]; float mx = -1e30f;
    for (int kt = 0; kt < 5; ++kt) {
      const _Float16* kv = QKV + kt * 3072 + m * 192 + 64 + h * 8;
      float s = 0.f;
#pragma unroll
      for (int d = 0; d < 8; ++d) s += (float)qv[d] * (float)kv[d];
      s *= 0.35355339059f;
      if (MSK[m * 8 + kt] != 0.f) s = -1e9f;
      sc[kt] = s; mx = fmaxf(mx, s);
    }
    float den = 0.f;
    for (int kt = 0; kt < 5; ++kt) { sc[kt] = expf(sc[kt] - mx); den += sc[kt]; }
    const float inv = 1.f / den;
#pragma unroll
    for (int d = 0; d < 8; ++d) {
      float o = 0.f;
      for (int kt = 0; kt < 5; ++kt)
        o += sc[kt] * (float)QKV[kt * 3072 + m * 192 + 128 + h * 8 + d];
      TMP[m * 64 + h * 8 + d] = (_Float16)(o * inv);
    }
  }
  // out-proj + residual into query token
  gemm16x64(TMP, Wo_c, P.c_out_b, 64, lane, TOK + 5 * 1024, 64, nullptr, 0);
  // cross ffn
  for (int it = lane; it < 16; it += 32)
    ln_to_f16(TOK + (5 * 16 + it) * 64, TMP + it * 64, P.c_fln_g, P.c_fln_b, true);
  gemm16x64(TMP, Wf_c, P.c_f_b, 64, lane, TOK + 5 * 1024, 64, nullptr, 0);

  // ================= HEAD: gelu(ln(q)) @ Wh^T + bh -> global =================
  for (int it = lane; it < 16; it += 32)
    ln_to_f16(TOK + (5 * 16 + it) * 64, TMP + it * 64, P.h_ln_g, P.h_ln_b, true);
  for (int n0 = 0; n0 < 32; n0 += 16) {
    v8f acc = {};
    {
      v16h a = ldA(TMP, 0, lane);
      v16h b = ldB(Whd, n0, 0, lane);
      acc = __builtin_amdgcn_wmma_f32_16x16x32_f16(false, a, false, b, (short)0, acc, false, false);
    }
    {
      v16h a = ldA(TMP, 32, lane);
      v16h b = ldB(Whd, n0, 32, lane);
      acc = __builtin_amdgcn_wmma_f32_16x16x32_f16(false, a, false, b, (short)0, acc, false, false);
    }
    const int   n  = n0 + (lane & 15);
    const float bv = P.h_b[n];
    const int   mh = (lane >> 4) * 8;
#pragma unroll
    for (int r = 0; r < 8; ++r) {
      const long long row = rowBase + mh + r;
      if (row < P.B) P.out[row * 32 + n] = acc[r] + bv;
    }
  }
}

extern "C" void kernel_launch(void* const* d_in, const int* in_sizes, int n_in,
                              void* d_out, int out_size, void* d_ws, size_t ws_size,
                              hipStream_t stream) {
  (void)n_in; (void)out_size; (void)d_ws; (void)ws_size;
  CondParams P;
  P.x = (const float*)d_in[0];
  P.inn_w = (const float*)d_in[1];   P.inn_b = (const float*)d_in[2];
  P.jet_w = (const float*)d_in[3];   P.jet_b = (const float*)d_in[4];
  P.dil_w = (const float*)d_in[5];   P.dil_b = (const float*)d_in[6];
  P.ang_w = (const float*)d_in[7];   P.ang_b = (const float*)d_in[8];
  P.s_ln_g = (const float*)d_in[9];  P.s_ln_b = (const float*)d_in[10];
  P.s_in_w = (const float*)d_in[11]; P.s_in_b = (const float*)d_in[12];
  P.s_out_w = (const float*)d_in[13];P.s_out_b = (const float*)d_in[14];
  P.s_fln_g = (const float*)d_in[15];P.s_fln_b = (const float*)d_in[16];
  P.s_f_w = (const float*)d_in[17];  P.s_f_b = (const float*)d_in[18];
  P.c_lnq_g = (const float*)d_in[19];P.c_lnq_b = (const float*)d_in[20];
  P.c_lnk_g = (const float*)d_in[21];P.c_lnk_b = (const float*)d_in[22];
  P.c_in_w = (const float*)d_in[23]; P.c_in_b = (const float*)d_in[24];
  P.c_out_w = (const float*)d_in[25];P.c_out_b = (const float*)d_in[26];
  P.c_fln_g = (const float*)d_in[27];P.c_fln_b = (const float*)d_in[28];
  P.c_f_w = (const float*)d_in[29];  P.c_f_b = (const float*)d_in[30];
  P.h_ln_g = (const float*)d_in[31]; P.h_ln_b = (const float*)d_in[32];
  P.h_w = (const float*)d_in[33];    P.h_b = (const float*)d_in[34];
  P.out = (float*)d_out;
  P.B = (long long)in_sizes[0] / 37;

  const int rowsPerBlock = 16 * N_WAVES;
  const int grid = (int)((P.B + rowsPerBlock - 1) / rowsPerBlock);
  condnet_fused<<<grid, 96, SMEM_TOTAL, stream>>>(P);
}
